// BiLSTM_CRF_66314295050415
// MI455X (gfx1250) — matrix operand order repaired and checked by reading, not compiled, not run on hardware
//
#include <hip/hip_runtime.h>
#include <hip/hip_bf16.h>
#include <math.h>

// ---------------------------------------------------------------------------
// BiLSTM-CRF for MI455X (gfx1250, wave32, WMMA + TDM).
//
// Pipeline:
//   1) conv feature extraction -> x[T,1728] (stored bf16 for WMMA GEMM)
//   2) fp32->bf16 weight conversion for the 4 input-projection matrices
//   3) gx = x @ wih^T + (bih+bhh) via v_wmma_f32_16x16x32_bf16; the B (weight)
//      tile is staged into LDS by the Tensor Data Mover (tensor_load_to_lds,
//      TENSORcnt double buffering); A streams via global_load + prefetch.
//   4) persistent recurrent scan kernels (fp32 GEMV per step, grid barrier)
//   5) h2t projection -> feats[T,27]
//   6) Viterbi scan + backtrack -> d_out = [score, path(8192)]
// ---------------------------------------------------------------------------

typedef __bf16 bf16;
typedef __attribute__((ext_vector_type(16))) __bf16 v16bf;
typedef __attribute__((ext_vector_type(8)))  float  v8f;
typedef unsigned int uintv4 __attribute__((ext_vector_type(4)));
typedef int          intv4  __attribute__((ext_vector_type(4)));
typedef int          intv8  __attribute__((ext_vector_type(8)));

#define T_STEPS 8192
#define D_IN    1728
#define HID     512
#define GATES   2048   // 4*HID
#define TAGS    27
#define START_T 25
#define STOP_T  26

#if __has_builtin(__builtin_amdgcn_tensor_load_to_lds)
#define USE_TDM 1
#else
#define USE_TDM 0
#endif

__device__ __forceinline__ float sigmoidf_(float x) {
    return 1.0f / (1.0f + __expf(-x));
}

// ---------------------------------------------------------------------------
// fp32 -> bf16 conversion (grid-stride)
// ---------------------------------------------------------------------------
__global__ void cvt_f32_bf16(const float* __restrict__ src,
                             bf16* __restrict__ dst, int n) {
    for (int i = blockIdx.x * blockDim.x + threadIdx.x; i < n;
         i += gridDim.x * blockDim.x) {
        dst[i] = (bf16)src[i];
    }
}

// ---------------------------------------------------------------------------
// Conv feature extraction: features[T,1,7,64] -> x[T,1728] (bf16)
// layout of the 1728: [16x1x12 conv1 | 16x4x12 conv3 | 16x4x12 conv5]
// ---------------------------------------------------------------------------
__global__ void conv_feats(const float* __restrict__ F,
                           const float* __restrict__ c1w, const float* __restrict__ c1b,
                           const float* __restrict__ c3w, const float* __restrict__ c3b,
                           const float* __restrict__ c5w, const float* __restrict__ c5b,
                           bf16* __restrict__ X) {
    long idx = (long)blockIdx.x * blockDim.x + threadIdx.x;
    const long total = (long)T_STEPS * D_IN;
    if (idx >= total) return;
    int t = (int)(idx / D_IN);
    int f = (int)(idx % D_IN);
    const float* ft = F + (long)t * 7 * 64;
    float acc;
    if (f < 192) {                       // conv1: [16,1,7,4], stride(1,4), W'=12
        int c = f / 12, ow = f % 12;
        acc = c1b[c];
        const float* w = c1w + c * 28;
        #pragma unroll
        for (int kh = 0; kh < 7; kh++)
            #pragma unroll
            for (int kw = 0; kw < 4; kw++)
                acc += ft[kh * 64 + ow * 4 + kw] * w[kh * 4 + kw];
    } else if (f < 960) {                // conv3: [16,1,4,12], H'=4, W'=12
        int r = f - 192;
        int c = r / 48, oh = (r % 48) / 12, ow = r % 12;
        acc = c3b[c];
        const float* w = c3w + c * 48;
        #pragma unroll
        for (int kh = 0; kh < 4; kh++)
            for (int kw = 0; kw < 12; kw++)
                acc += ft[(oh + kh) * 64 + ow * 4 + kw] * w[kh * 12 + kw];
    } else {                             // conv5: [16,1,4,20], H'=4, W'=12
        int r = f - 960;
        int c = r / 48, oh = (r % 48) / 12, ow = r % 12;
        acc = c5b[c];
        const float* w = c5w + c * 80;
        #pragma unroll
        for (int kh = 0; kh < 4; kh++)
            for (int kw = 0; kw < 20; kw++)
                acc += ft[(oh + kh) * 64 + ow * 4 + kw] * w[kh * 20 + kw];
    }
    X[idx] = (bf16)acc;
}

// ---------------------------------------------------------------------------
// TDM: issue a 2D tensor tile load Global->LDS.
// Tile: 512 rows x 32 bf16 columns of W[N,K] starting at `gaddr`
// (rows separated by K elements), packed contiguously in LDS (row*64B).
// Descriptor packing per CDNA5 ISA 8.3/8.4 (groups 0/1; groups 2/3 unused).
// ---------------------------------------------------------------------------
#if USE_TDM
__device__ __forceinline__ void tdm_load_wtile(const bf16* gaddr,
                                               unsigned lds_off, int K, int N) {
    unsigned long long ga = (unsigned long long)gaddr;
    uintv4 g0;
    g0[0] = 1u;                                   // count=1, user descriptor
    g0[1] = lds_off;                              // LDS byte address
    g0[2] = (unsigned)(ga & 0xFFFFFFFFu);         // global_addr[31:0]
    g0[3] = (unsigned)(((ga >> 32) & 0x1FFFFFFu)  // global_addr[56:32]
                       | (2u << 30));             // type=2 ("image")
    intv8 g1;
    g1[0] = (int)(1u << 16);                      // data_size=1 -> 2 bytes
    g1[1] = (int)((unsigned)(K & 0xFFFF) << 16);  // tensor_dim0[15:0]
    g1[2] = (int)(((unsigned)(K >> 16) & 0xFFFFu) // tensor_dim0[31:16]
                  | ((unsigned)(N & 0xFFFF) << 16));      // tensor_dim1[15:0]
    g1[3] = (int)(((unsigned)(N >> 16) & 0xFFFFu)         // tensor_dim1[31:16]
                  | (32u << 16));                 // tile_dim0 = 32 elements
    g1[4] = (int)512;                             // tile_dim1 = 512 rows
    g1[5] = (int)K;                               // tensor_dim0_stride[31:0]
    g1[6] = 0;                                    // stride0 hi / stride1 lo
    g1[7] = 0;                                    // stride1 hi
    intv4 gz = {0, 0, 0, 0};
#if defined(__clang_major__) && (__clang_major__ >= 23)
    intv8 gz8 = {0, 0, 0, 0, 0, 0, 0, 0};
    __builtin_amdgcn_tensor_load_to_lds(g0, g1, gz, gz, gz8, 0);
#else
    __builtin_amdgcn_tensor_load_to_lds(g0, g1, gz, gz, 0);
#endif
}
#endif

// ---------------------------------------------------------------------------
// WMMA GEMM: C[M,N] = A[M,K](bf16) * W[N,K](bf16)^T + (bih+bhh)[N]
// One block = 256 threads = 8 waves. Block tile: 16 rows x 512 cols
// (each wave owns 16x64 via 4 accumulators). K stepped by 32.
// B tile (512 rows x 32 K) staged to LDS by TDM (wave 0 is the producer),
// double-buffered on TENSORcnt. A streams directly from global.
//
// Fragment layouts used here (wave32):
//  A 16x32 bf16 : lane holds row M=lane&15; lane half selects K range (0..15 / 16..31);
//                 VGPR v -> K = (lane>>4)*16 + v*2 (+0,+1)
//  B 32x16 bf16 : lane holds col N=lane&15; lane half selects K range
//                 (0..15 / 16..31), 16 contiguous K per lane
//  C/D 16x16 f32: VGPR r -> M = r + 8*(lane>>4), N = lane&15
// ---------------------------------------------------------------------------
__global__ void __launch_bounds__(256)
gemm_bf16_wmma(const bf16* __restrict__ A,   // [M,K]
               const bf16* __restrict__ W,   // [N,K]
               const float* __restrict__ bih,
               const float* __restrict__ bhh,
               float* __restrict__ C, int M, int N, int K) {
    __shared__ __align__(16) bf16 smem[2][512 * 32];   // 2 x 32KB W tiles

    const int lane = threadIdx.x & 31;
    const int wave = threadIdx.x >> 5;
    const int m0 = blockIdx.x * 16;
    const int nblk = blockIdx.y * 512;               // block's W row range
    const int n0 = nblk + wave * 64;

    const int rowA = m0 + (lane & 15);
    const int kkA  = (lane >> 4) * 16;               // A: lane-half K offset
    const int nl   = lane & 15;
    const int kbB  = (lane >> 4) * 16;               // B: lane-half K offset
    const int colLoc = wave * 64 + nl;               // tile-local B row

    const bf16* __restrict__ Arow = A + (size_t)rowA * K;

    v8f acc0 = {}, acc1 = {}, acc2 = {}, acc3 = {};

    const int nch = K / 32;
#if USE_TDM
    const unsigned lds0 = (unsigned)(size_t)(&smem[0][0]);
    const unsigned lds1 = (unsigned)(size_t)(&smem[1][0]);
    if (wave == 0) {
        tdm_load_wtile(W + (size_t)nblk * K, lds0, K, N);
    }
#endif

    for (int i = 0; i < nch; i++) {
        const int k0 = i * 32;
        const int cur = i & 1;
        __syncthreads();          // buf[cur^1] readers from iter i-1 are done
#if USE_TDM
        if (wave == 0) {
            if (i + 1 < nch) {
                tdm_load_wtile(W + (size_t)nblk * K + (i + 1) * 32,
                               (i & 1) ? lds0 : lds1, K, N);
                __builtin_amdgcn_s_wait_tensorcnt(1);   // buf[cur] complete
            } else {
                __builtin_amdgcn_s_wait_tensorcnt(0);
            }
        }
#else
        // Fallback: cooperative staging of the current W chunk.
        for (int e = threadIdx.x; e < 512 * 32; e += 256) {
            int r = e >> 5, c = e & 31;
            smem[cur][e] = W[(size_t)(nblk + r) * K + k0 + c];
        }
#endif
        __syncthreads();          // buf[cur] visible to all waves

        __builtin_prefetch(Arow + k0 + 128, 0, 1);     // global_prefetch_b8

        v16bf a;
        #pragma unroll
        for (int v = 0; v < 8; v++) {
            int kb = k0 + kkA + (v << 1);
            a[2 * v]     = Arow[kb];
            a[2 * v + 1] = Arow[kb + 1];
        }

        const bf16* __restrict__ smrow = &smem[cur][colLoc * 32 + kbB];
        v16bf b0, b1, b2, b3;
        #pragma unroll
        for (int e = 0; e < 16; e++) {
            b0[e] = smrow[e];              // cols n0 +  0..15
            b1[e] = smrow[512 + e];        // cols n0 + 16..31 (16 rows * 32)
            b2[e] = smrow[1024 + e];       // cols n0 + 32..47
            b3[e] = smrow[1536 + e];       // cols n0 + 48..63
        }
        acc0 = __builtin_amdgcn_wmma_f32_16x16x32_bf16(false, a, false, b0,
                                                       (short)0, acc0, false, false);
        acc1 = __builtin_amdgcn_wmma_f32_16x16x32_bf16(false, a, false, b1,
                                                       (short)0, acc1, false, false);
        acc2 = __builtin_amdgcn_wmma_f32_16x16x32_bf16(false, a, false, b2,
                                                       (short)0, acc2, false, false);
        acc3 = __builtin_amdgcn_wmma_f32_16x16x32_bf16(false, a, false, b3,
                                                       (short)0, acc3, false, false);
    }

    const int nh = lane >> 4;
    #pragma unroll
    for (int r = 0; r < 8; r++) {
        int m = m0 + r + 8 * nh;
        float* Crow = C + (size_t)m * N;
        int n;
        n = n0 +  0 + nl; Crow[n] = acc0[r] + bih[n] + bhh[n];
        n = n0 + 16 + nl; Crow[n] = acc1[r] + bih[n] + bhh[n];
        n = n0 + 32 + nl; Crow[n] = acc2[r] + bih[n] + bhh[n];
        n = n0 + 48 + nl; Crow[n] = acc3[r] + bih[n] + bhh[n];
    }
}

// ---------------------------------------------------------------------------
// Persistent recurrent LSTM scan (one direction).
// Grid = 32 blocks x 64 threads. Block b owns hidden units j in [16b,16b+16).
// Thread t = gate*16 + jl computes one gate dot-product of length 512.
// h double-buffered in global (hbuf[2][512]); agent-scope atomic grid barrier
// per step. c-state lives in registers of threads 0..15.
// ---------------------------------------------------------------------------
__global__ void __launch_bounds__(64)
lstm_recurrent(const float* __restrict__ gx,   // [T, 2048]
               const float* __restrict__ whh,  // [2048, 512] fp32
               float* __restrict__ hs,         // [T, 1024]
               float* __restrict__ hbuf,       // [2][512]
               int* __restrict__ sync,
               int forward, int colbase) {
    const int t  = threadIdx.x;        // 0..63
    const int b  = t >> 4;             // gate block 0..3 (i,f,g,o)
    const int jl = t & 15;
    const int j  = blockIdx.x * 16 + jl;
    const int grow = b * HID + j;      // row of whh / gx column
    const float4* __restrict__ w4 =
        (const float4*)(whh + (size_t)grow * HID);

    __shared__ float gsm[4][16];
    float cst = 0.0f;                  // cell state (threads 0..15 only)

    for (int step = 0; step < T_STEPS; step++) {
        const int ta = forward ? step : (T_STEPS - 1 - step);
        const float4* __restrict__ h4 =
            (const float4*)(hbuf + (size_t)(step & 1) * HID);

        float acc = gx[(size_t)ta * GATES + grow];
        #pragma unroll 8
        for (int k = 0; k < HID / 4; k++) {
            float4 hv = h4[k];
            float4 wv = w4[k];
            acc += wv.x * hv.x + wv.y * hv.y + wv.z * hv.z + wv.w * hv.w;
        }
        gsm[b][jl] = acc;
        __syncthreads();

        if (t < 16) {
            float ig = sigmoidf_(gsm[0][t]);
            float fg = sigmoidf_(gsm[1][t]);
            float gg = tanhf(gsm[2][t]);
            float og = sigmoidf_(gsm[3][t]);
            cst = fg * cst + ig * gg;
            float hn = og * tanhf(cst);
            hs[(size_t)ta * (2 * HID) + colbase + blockIdx.x * 16 + t] = hn;
            hbuf[(size_t)((step + 1) & 1) * HID + blockIdx.x * 16 + t] = hn;
        }
        __threadfence();               // make hbuf visible device-wide

        // grid barrier (agent scope acquire/release)
        __syncthreads();
        if (t == 0) {
            int g = __hip_atomic_load(&sync[1], __ATOMIC_RELAXED,
                                      __HIP_MEMORY_SCOPE_AGENT);
            int arrived = __hip_atomic_fetch_add(&sync[0], 1, __ATOMIC_ACQ_REL,
                                                 __HIP_MEMORY_SCOPE_AGENT);
            if (arrived == (int)gridDim.x - 1) {
                __hip_atomic_store(&sync[0], 0, __ATOMIC_RELAXED,
                                   __HIP_MEMORY_SCOPE_AGENT);
                __hip_atomic_store(&sync[1], g + 1, __ATOMIC_RELEASE,
                                   __HIP_MEMORY_SCOPE_AGENT);
            } else {
                while (__hip_atomic_load(&sync[1], __ATOMIC_ACQUIRE,
                                         __HIP_MEMORY_SCOPE_AGENT) == g) {
                    __builtin_amdgcn_s_sleep(1);
                }
            }
        }
        __syncthreads();
    }
}

// ---------------------------------------------------------------------------
// h2t projection: feats[T,27] = hs1[T,1024] @ h2t_w[27,1024]^T + h2t_b
// ---------------------------------------------------------------------------
__global__ void h2t_proj(const float* __restrict__ hs,
                         const float* __restrict__ w,
                         const float* __restrict__ bias,
                         float* __restrict__ feats) {
    int idx = blockIdx.x * blockDim.x + threadIdx.x;
    if (idx >= T_STEPS * TAGS) return;
    int t = idx / TAGS, tag = idx % TAGS;
    const float* h = hs + (size_t)t * (2 * HID);
    const float* wr = w + (size_t)tag * (2 * HID);
    float acc = bias[tag];
    #pragma unroll 8
    for (int k = 0; k < 2 * HID; k++) acc += h[k] * wr[k];
    feats[idx] = acc;
}

// ---------------------------------------------------------------------------
// Viterbi: forward scan (argmax/backpointers) + backtrack.
// One wave; fv kept in LDS. out[0]=score, out[1..T]=path.
// ---------------------------------------------------------------------------
__global__ void viterbi(const float* __restrict__ feats,
                        const float* __restrict__ trans,
                        int* __restrict__ bptr,
                        float* __restrict__ out) {
    __shared__ float fv[32];
    __shared__ float nf[32];
    const int t = threadIdx.x;
    if (t < TAGS) fv[t] = (t == START_T) ? 0.0f : -10000.0f;
    __syncthreads();

    for (int step = 0; step < T_STEPS; step++) {
        if (t < TAGS) {
            float best = -3.4e38f;
            int bi = 0;
            for (int p = 0; p < TAGS; p++) {
                float s = fv[p] + trans[t * TAGS + p];
                if (s > best) { best = s; bi = p; }
            }
            bptr[step * TAGS + t] = bi;
            nf[t] = best + feats[step * TAGS + t];
        }
        __syncthreads();
        if (t < TAGS) fv[t] = nf[t];
        __syncthreads();
    }

    if (t == 0) {
        float best = -3.4e38f;
        int bl = 0;
        for (int p = 0; p < TAGS; p++) {
            float s = fv[p] + trans[STOP_T * TAGS + p];
            if (s > best) { best = s; bl = p; }
        }
        out[0] = best;
        int tag = bl;
        for (int i = T_STEPS - 1; i >= 0; i--) {
            out[1 + i] = (float)tag;
            tag = bptr[i * TAGS + tag];
        }
    }
}

// ---------------------------------------------------------------------------
// Host-side orchestration
// ---------------------------------------------------------------------------
extern "C" void kernel_launch(void* const* d_in, const int* in_sizes, int n_in,
                              void* d_out, int out_size, void* d_ws, size_t ws_size,
                              hipStream_t stream) {
    const float* features = (const float*)d_in[0];
    const float* c1w = (const float*)d_in[1];
    const float* c1b = (const float*)d_in[2];
    const float* c3w = (const float*)d_in[3];
    const float* c3b = (const float*)d_in[4];
    const float* c5w = (const float*)d_in[5];
    const float* c5b = (const float*)d_in[6];
    const float* wih0f = (const float*)d_in[7];
    const float* whh0f = (const float*)d_in[8];
    const float* bih0f = (const float*)d_in[9];
    const float* bhh0f = (const float*)d_in[10];
    const float* wih0r = (const float*)d_in[11];
    const float* whh0r = (const float*)d_in[12];
    const float* bih0r = (const float*)d_in[13];
    const float* bhh0r = (const float*)d_in[14];
    const float* wih1f = (const float*)d_in[15];
    const float* whh1f = (const float*)d_in[16];
    const float* bih1f = (const float*)d_in[17];
    const float* bhh1f = (const float*)d_in[18];
    const float* wih1r = (const float*)d_in[19];
    const float* whh1r = (const float*)d_in[20];
    const float* bih1r = (const float*)d_in[21];
    const float* bhh1r = (const float*)d_in[22];
    const float* h2t_w = (const float*)d_in[23];
    const float* h2t_b = (const float*)d_in[24];
    const float* trans = (const float*)d_in[25];

    char* ws = (char*)d_ws;
    size_t off = 0;
    auto alloc = [&](size_t bytes) -> size_t {
        size_t o = off;
        off = (off + bytes + 255) & ~(size_t)255;
        return o;
    };
    const size_t oX0  = alloc((size_t)T_STEPS * D_IN * 2);       // x bf16
    const size_t oW0F = alloc((size_t)GATES * D_IN * 2);
    const size_t oW0R = alloc((size_t)GATES * D_IN * 2);
    const size_t oW1F = alloc((size_t)GATES * 2 * HID * 2);
    const size_t oW1R = alloc((size_t)GATES * 2 * HID * 2);
    const size_t oGXF = alloc((size_t)T_STEPS * GATES * 4);
    const size_t oGXR = alloc((size_t)T_STEPS * GATES * 4);
    const size_t oHS0 = alloc((size_t)T_STEPS * 2 * HID * 4);
    const size_t oHS1 = alloc((size_t)T_STEPS * 2 * HID * 4);
    const size_t oX2  = alloc((size_t)T_STEPS * 2 * HID * 2);    // x2 bf16
    const size_t oFE  = alloc((size_t)T_STEPS * TAGS * 4);
    const size_t oBP  = alloc((size_t)T_STEPS * TAGS * 4);
    const size_t oSY  = alloc(32 + 2 * HID * 4);                 // sync + hbuf
    (void)ws_size; (void)in_sizes; (void)n_in; (void)out_size;

    bf16* x0bf = (bf16*)(ws + oX0);
    bf16* w0f  = (bf16*)(ws + oW0F);
    bf16* w0r  = (bf16*)(ws + oW0R);
    bf16* w1f  = (bf16*)(ws + oW1F);
    bf16* w1r  = (bf16*)(ws + oW1R);
    float* gxf = (float*)(ws + oGXF);
    float* gxr = (float*)(ws + oGXR);
    float* hs0 = (float*)(ws + oHS0);
    float* hs1 = (float*)(ws + oHS1);
    bf16* x2bf = (bf16*)(ws + oX2);
    float* fe  = (float*)(ws + oFE);
    int*   bp  = (int*)(ws + oBP);
    int*   sy  = (int*)(ws + oSY);
    float* hbuf = (float*)(ws + oSY + 32);
    const size_t syncBytes = 32 + 2 * HID * 4;

    // 1) weight conversions
    {
        int n0 = GATES * D_IN;
        int n1 = GATES * 2 * HID;
        cvt_f32_bf16<<<(n0 + 255) / 256, 256, 0, stream>>>(wih0f, w0f, n0);
        cvt_f32_bf16<<<(n0 + 255) / 256, 256, 0, stream>>>(wih0r, w0r, n0);
        cvt_f32_bf16<<<(n1 + 255) / 256, 256, 0, stream>>>(wih1f, w1f, n1);
        cvt_f32_bf16<<<(n1 + 255) / 256, 256, 0, stream>>>(wih1r, w1r, n1);
    }

    // 2) conv features -> x bf16
    {
        long total = (long)T_STEPS * D_IN;
        conv_feats<<<(unsigned)((total + 255) / 256), 256, 0, stream>>>(
            features, c1w, c1b, c3w, c3b, c5w, c5b, x0bf);
    }

    // 3) layer-0 input projections (WMMA + TDM staging)
    {
        dim3 grid(T_STEPS / 16, GATES / 512);
        gemm_bf16_wmma<<<grid, 256, 0, stream>>>(x0bf, w0f, bih0f, bhh0f, gxf,
                                                 T_STEPS, GATES, D_IN);
        gemm_bf16_wmma<<<grid, 256, 0, stream>>>(x0bf, w0r, bih0r, bhh0r, gxr,
                                                 T_STEPS, GATES, D_IN);
    }

    // 4) layer-0 recurrent scans
    hipMemsetAsync(ws + oSY, 0, syncBytes, stream);
    lstm_recurrent<<<HID / 16, 64, 0, stream>>>(gxf, whh0f, hs0, hbuf, sy, 1, 0);
    hipMemsetAsync(ws + oSY, 0, syncBytes, stream);
    lstm_recurrent<<<HID / 16, 64, 0, stream>>>(gxr, whh0r, hs0, hbuf, sy, 0, HID);

    // 5) hs0 -> bf16 for layer-1 GEMMs
    {
        int n = T_STEPS * 2 * HID;
        cvt_f32_bf16<<<(n + 255) / 256, 256, 0, stream>>>(hs0, x2bf, n);
    }

    // 6) layer-1 input projections (WMMA + TDM staging, K=1024)
    {
        dim3 grid(T_STEPS / 16, GATES / 512);
        gemm_bf16_wmma<<<grid, 256, 0, stream>>>(x2bf, w1f, bih1f, bhh1f, gxf,
                                                 T_STEPS, GATES, 2 * HID);
        gemm_bf16_wmma<<<grid, 256, 0, stream>>>(x2bf, w1r, bih1r, bhh1r, gxr,
                                                 T_STEPS, GATES, 2 * HID);
    }

    // 7) layer-1 recurrent scans
    hipMemsetAsync(ws + oSY, 0, syncBytes, stream);
    lstm_recurrent<<<HID / 16, 64, 0, stream>>>(gxf, whh1f, hs1, hbuf, sy, 1, 0);
    hipMemsetAsync(ws + oSY, 0, syncBytes, stream);
    lstm_recurrent<<<HID / 16, 64, 0, stream>>>(gxr, whh1r, hs1, hbuf, sy, 0, HID);

    // 8) tag projection
    {
        int n = T_STEPS * TAGS;
        h2t_proj<<<(n + 255) / 256, 256, 0, stream>>>(hs1, h2t_w, h2t_b, fe);
    }

    // 9) Viterbi
    viterbi<<<1, 32, 0, stream>>>(fe, trans, bp, (float*)d_out);
}
